// SparseSubdivideBlock3d_68169720922159
// MI455X (gfx1250) — matrix operand order repaired and checked by reading, not compiled, not run on hardware
//
#include <hip/hip_runtime.h>
#include <cmath>

// ---------- types ----------
typedef _Float16 v16h __attribute__((ext_vector_type(16)));
typedef _Float16 v8h  __attribute__((ext_vector_type(8)));
typedef float    v8f  __attribute__((ext_vector_type(8)));

#define CIN   128
#define COUT  64
#define NBAT  2
#define RES   32
#define NPB   16384
#define NPTS  32768      // NBAT*NPB coarse points
#define NF    262144     // 8*NPTS fine points
#define EPSV  1e-5f

static __device__ __forceinline__ v8f wmma_f16(v16h a, v16h b, v8f c) {
  return __builtin_amdgcn_wmma_f32_16x16x32_f16(false, a, false, b, (short)0, c,
                                                false, false);
}
static __device__ __forceinline__ float silu(float x) {
  return x / (1.0f + expf(-x));
}

// per-dim tap set for aggregated conv1 weights: (child bit od, tap select sd)
// returns kernel indices kd (= d+1) contributing to coarse neighbor e = od+sd-1
static __device__ __forceinline__ int tapset(int od, int sd, int* l) {
  if (sd == 0) { if (od == 0) { l[0] = 0; return 1; } l[0] = 0; l[1] = 1; return 2; }
  if (od == 0) { l[0] = 1; l[1] = 2; return 2; }
  l[0] = 2; return 1;
}

// ---------- weight preparation ----------
// W1agg[o][s][n][k] (f16, n-major / k-contiguous == WMMA B-friendly)
__global__ void k_w1agg(const float* __restrict__ w1, _Float16* __restrict__ w1aggT) {
  int t = blockIdx.x * 256 + threadIdx.x;          // 64*8192 = 524288
  int mat = t >> 13; int rem = t & 8191;
  int n = rem >> 7;  int k = rem & 127;
  int o = mat >> 3;  int s = mat & 7;
  int lx[2], ly[2], lz[2];
  int cx = tapset((o >> 2) & 1, (s >> 2) & 1, lx);
  int cy = tapset((o >> 1) & 1, (s >> 1) & 1, ly);
  int cz = tapset(o & 1, s & 1, lz);
  float sum = 0.0f;
  for (int a = 0; a < cx; a++)
    for (int b = 0; b < cy; b++)
      for (int c = 0; c < cz; c++)
        sum += w1[(size_t)(((lx[a] * 3 + ly[b]) * 3 + lz[c])) * (CIN * COUT) +
                  (size_t)k * COUT + n];
  w1aggT[((size_t)mat * COUT + n) * CIN + k] = (_Float16)sum;
}

// w2T[kd][n][k] f16
__global__ void k_w2t(const float* __restrict__ w2, _Float16* __restrict__ w2T) {
  int t = blockIdx.x * 256 + threadIdx.x;          // 27*4096 = 110592
  int kd = t >> 12; int n = (t >> 6) & 63; int k = t & 63;
  w2T[((size_t)kd * COUT + n) * COUT + k] =
      (_Float16)w2[(size_t)kd * (COUT * COUT) + (size_t)k * COUT + n];
}

// wskipT[n][k] f16
__global__ void k_wskipt(const float* __restrict__ ws_, _Float16* __restrict__ wT) {
  int t = blockIdx.x * 256 + threadIdx.x;          // 8192
  int n = t >> 7; int k = t & 127;
  wT[(size_t)n * CIN + k] = (_Float16)ws_[(size_t)k * COUT + n];
}

// ---------- coarse voxel table ----------
__global__ void k_tinit(int* __restrict__ table) {
  table[blockIdx.x * 256 + threadIdx.x] = -1;      // NBAT*32^3 = 65536
}
__global__ void k_tscat(const int* __restrict__ coords, int* __restrict__ table) {
  int i = blockIdx.x * 256 + threadIdx.x;          // NPTS
  int b = coords[4 * i + 0], x = coords[4 * i + 1];
  int y = coords[4 * i + 2], z = coords[4 * i + 3];
  table[((b * RES + x) * RES + y) * RES + z] = i;
}

// ---------- GroupNorm 1 (C=128, G=32, per batch) ----------
__global__ void k_gn1_part(const float* __restrict__ feats, float* __restrict__ part) {
  __shared__ float s0[256], s1[256];
  int t = threadIdx.x, blk = blockIdx.x;           // 256 blocks, 128 rows each
  int c = t & 127, h = t >> 7;
  int rbase = blk * 128;
  float sum = 0.f, ss = 0.f;
  for (int k = 0; k < 64; k++) {
    int r = rbase + h + 2 * k;
    float v = feats[(size_t)r * CIN + c];
    sum += v; ss += v * v;
  }
  s0[t] = sum; s1[t] = ss; __syncthreads();
  if (t < 128) {
    size_t i = ((size_t)blk * 128 + t) * 2;
    part[i]     = s0[t] + s0[t + 128];
    part[i + 1] = s1[t] + s1[t + 128];
  }
}
__global__ void k_gn1_final(const float* __restrict__ part,
                            const float* __restrict__ gw, const float* __restrict__ gb,
                            float* __restrict__ gAB) {
  __shared__ float cs[256], css[256], mv[64][2];
  int t = threadIdx.x; int b = t >> 7; int c = t & 127;
  float s = 0.f, ss = 0.f;
  for (int blk = 0; blk < 128; blk++) {
    size_t i = (((size_t)(b * 128 + blk)) * 128 + c) * 2;
    s += part[i]; ss += part[i + 1];
  }
  cs[t] = s; css[t] = ss; __syncthreads();
  if (t < 64) {
    int bb = t >> 5, g = t & 31;
    float S = 0.f, SS = 0.f;
    for (int q = 0; q < 4; q++) { int i = bb * 128 + g * 4 + q; S += cs[i]; SS += css[i]; }
    float cnt = 16384.0f * 4.0f;
    float mean = S / cnt;
    mv[t][0] = mean; mv[t][1] = SS / cnt - mean * mean;
  }
  __syncthreads();
  int g = c >> 2;
  float mean = mv[b * 32 + g][0], var = mv[b * 32 + g][1];
  float A = gw[c] * rsqrtf(var + EPSV);
  float Bc = gb[c] - mean * A;
  gAB[((size_t)(b * 128 + c)) * 2] = A;
  gAB[((size_t)(b * 128 + c)) * 2 + 1] = Bc;
}

// apply GN1 + SiLU -> hcoarse(f16); also raw feats -> f16 (for skip GEMM)
__global__ void k_apply1(const float* __restrict__ feats, const float* __restrict__ gAB,
                         _Float16* __restrict__ f16raw, _Float16* __restrict__ hcoarse) {
  int base = (blockIdx.x * 256 + threadIdx.x) * 4;  // 4096 blocks -> 4.19M elems
#pragma unroll
  for (int e = 0; e < 4; e++) {
    int idx = base + e;
    int r = idx >> 7, c = idx & 127, b = r >> 14;
    float v = feats[idx];
    f16raw[idx] = (_Float16)v;
    float A = gAB[((size_t)(b * 128 + c)) * 2];
    float Bc = gAB[((size_t)(b * 128 + c)) * 2 + 1];
    float x = A * v + Bc;
    hcoarse[idx] = (_Float16)silu(x);
  }
}

// ---------- conv1: 8 aggregated coarse taps, 128->64, WMMA f16 ----------
__global__ __launch_bounds__(128) void k_conv1(
    const _Float16* __restrict__ hcoarse, const int4* __restrict__ coords,
    const int* __restrict__ table, const _Float16* __restrict__ w1aggT,
    const float* __restrict__ b1, _Float16* __restrict__ h1) {
  int wid = blockIdx.x * 4 + (threadIdx.x >> 5);
  int lane = threadIdx.x & 31;
  int o = wid & 7; int pbase = (wid >> 3) * 16;
  int m = lane & 15, hi = lane >> 4;
  int4 cc = coords[pbase + m];
  int ox = (o >> 2) & 1, oy = (o >> 1) & 1, oz = o & 1;
  float bv[4];
#pragma unroll
  for (int nb = 0; nb < 4; nb++) bv[nb] = b1[nb * 16 + m];
  v8f acc[4] = {};
  for (int s = 0; s < 8; s++) {
    int nx = cc.y + ox + ((s >> 2) & 1) - 1;
    int ny = cc.z + oy + ((s >> 1) & 1) - 1;
    int nz = cc.w + oz + (s & 1) - 1;
    int j = -1;
    if (((unsigned)nx < 32u) && ((unsigned)ny < 32u) && ((unsigned)nz < 32u))
      j = table[((cc.x * RES + nx) * RES + ny) * RES + nz];
    const _Float16* rp = hcoarse + (size_t)j * CIN;
    const _Float16* wb = w1aggT + (size_t)(o * 8 + s) * COUT * CIN;
#pragma unroll
    for (int ks = 0; ks < 4; ks++) {
      union { v16h v; v8h h[2]; } A;
      v8h z = {}; A.h[0] = z; A.h[1] = z;
      int k0 = ks * 32 + hi * 16;
      if (j >= 0) {
        A.h[0] = *(const v8h*)(rp + k0);
        A.h[1] = *(const v8h*)(rp + k0 + 8);
      }
#pragma unroll
      for (int nb = 0; nb < 4; nb++) {
        v16h Bm = *(const v16h*)(wb + (size_t)(nb * 16 + m) * CIN + ks * 32 + hi * 16);
        acc[nb] = wmma_f16(A.v, Bm, acc[nb]);
      }
    }
  }
#pragma unroll
  for (int nb = 0; nb < 4; nb++)
#pragma unroll
    for (int v = 0; v < 8; v++) {
      int R = 8 * (pbase + v + 8 * hi) + o;
      h1[(size_t)R * COUT + nb * 16 + m] = (_Float16)(acc[nb][v] + bv[nb]);
    }
}

// ---------- GroupNorm 2 (C=64, G=32, per batch, over h1) ----------
__global__ void k_gn2_part(const _Float16* __restrict__ h1, float* __restrict__ part) {
  __shared__ float s0[256], s1[256];
  int t = threadIdx.x, blk = blockIdx.x;           // 512 blocks, 512 rows each
  int c = t & 63, ro = t >> 6;
  int rbase = blk * 512;
  float sum = 0.f, ss = 0.f;
  for (int k = 0; k < 128; k++) {
    int r = rbase + ro + 4 * k;
    float v = (float)h1[(size_t)r * COUT + c];
    sum += v; ss += v * v;
  }
  s0[t] = sum; s1[t] = ss; __syncthreads();
  if (t < 64) {
    float a = s0[t] + s0[t + 64] + s0[t + 128] + s0[t + 192];
    float b = s1[t] + s1[t + 64] + s1[t + 128] + s1[t + 192];
    size_t i = ((size_t)blk * 64 + t) * 2;
    part[i] = a; part[i + 1] = b;
  }
}
__global__ void k_gn2_final(const float* __restrict__ part,
                            const float* __restrict__ gw, const float* __restrict__ gb,
                            float* __restrict__ gAB) {
  __shared__ float cs[128], css[128], mv[64][2];
  int t = threadIdx.x; int b = t >> 6; int c = t & 63;   // 128 threads
  float s = 0.f, ss = 0.f;
  for (int blk = 0; blk < 256; blk++) {
    size_t i = (((size_t)(b * 256 + blk)) * 64 + c) * 2;
    s += part[i]; ss += part[i + 1];
  }
  cs[t] = s; css[t] = ss; __syncthreads();
  if (t < 64) {
    int bb = t >> 5, g = t & 31;
    float S = cs[bb * 64 + 2 * g] + cs[bb * 64 + 2 * g + 1];
    float SS = css[bb * 64 + 2 * g] + css[bb * 64 + 2 * g + 1];
    float cnt = 131072.0f * 2.0f;
    float mean = S / cnt;
    mv[t][0] = mean; mv[t][1] = SS / cnt - mean * mean;
  }
  __syncthreads();
  int g = c >> 1;
  float mean = mv[b * 32 + g][0], var = mv[b * 32 + g][1];
  float A = gw[c] * rsqrtf(var + EPSV);
  float Bc = gb[c] - mean * A;
  gAB[((size_t)(b * 64 + c)) * 2] = A;
  gAB[((size_t)(b * 64 + c)) * 2 + 1] = Bc;
}

__global__ void k_apply2(const _Float16* __restrict__ h1, const float* __restrict__ gAB,
                         _Float16* __restrict__ hmid) {
  int base = (blockIdx.x * 256 + threadIdx.x) * 4;  // 16384 blocks -> 16.78M elems
#pragma unroll
  for (int e = 0; e < 4; e++) {
    int idx = base + e;
    int r = idx >> 6, c = idx & 63, b = r >> 17;
    float A = gAB[((size_t)(b * 64 + c)) * 2];
    float Bc = gAB[((size_t)(b * 64 + c)) * 2 + 1];
    float x = A * (float)h1[idx] + Bc;
    hmid[idx] = (_Float16)silu(x);
  }
}

// ---------- conv2 (27 taps, 64->64) + skip GEMM (128->64), WMMA f16 ----------
__global__ __launch_bounds__(128) void k_conv2(
    const _Float16* __restrict__ hmid, const _Float16* __restrict__ f16raw,
    const int4* __restrict__ coords, const int* __restrict__ table,
    const _Float16* __restrict__ w2T, const _Float16* __restrict__ wskipT,
    const float* __restrict__ b2, const float* __restrict__ bskip,
    float* __restrict__ out) {
  int wid = blockIdx.x * 4 + (threadIdx.x >> 5);
  int lane = threadIdx.x & 31;
  int o = wid & 7; int pbase = (wid >> 3) * 16;
  int m = lane & 15, hi = lane >> 4;
  int4 cc = coords[pbase + m];
  int ox = (o >> 2) & 1, oy = (o >> 1) & 1, oz = o & 1;
  v8f acc[4] = {};
  for (int kd = 0; kd < 3; kd++)
    for (int ky = 0; ky < 3; ky++)
      for (int kz = 0; kz < 3; kz++) {
        int tx = ox + kd - 1, ty = oy + ky - 1, tz = oz + kz - 1;
        int ex = (tx < 0) ? -1 : (tx >> 1);
        int ey = (ty < 0) ? -1 : (ty >> 1);
        int ez = (tz < 0) ? -1 : (tz >> 1);
        int oi = (tx & 1) * 4 + (ty & 1) * 2 + (tz & 1);
        int nx = cc.y + ex, ny = cc.z + ey, nz = cc.w + ez;
        int j = -1;
        if (((unsigned)nx < 32u) && ((unsigned)ny < 32u) && ((unsigned)nz < 32u))
          j = table[((cc.x * RES + nx) * RES + ny) * RES + nz];
        const _Float16* rp = hmid + ((size_t)j * 8 + oi) * COUT;
        const _Float16* wb = w2T + (size_t)((kd * 3 + ky) * 3 + kz) * COUT * COUT;
#pragma unroll
        for (int ks = 0; ks < 2; ks++) {
          union { v16h v; v8h h[2]; } A;
          v8h z = {}; A.h[0] = z; A.h[1] = z;
          int k0 = ks * 32 + hi * 16;
          if (j >= 0) {
            A.h[0] = *(const v8h*)(rp + k0);
            A.h[1] = *(const v8h*)(rp + k0 + 8);
          }
#pragma unroll
          for (int nb = 0; nb < 4; nb++) {
            v16h Bm = *(const v16h*)(wb + (size_t)(nb * 16 + m) * COUT + ks * 32 + hi * 16);
            acc[nb] = wmma_f16(A.v, Bm, acc[nb]);
          }
        }
      }
  // skip: f16raw[parent] @ w_skip  (same for all 8 children of a parent)
  {
    const _Float16* rp = f16raw + (size_t)(pbase + m) * CIN;
#pragma unroll
    for (int ks = 0; ks < 4; ks++) {
      union { v16h v; v8h h[2]; } A;
      int k0 = ks * 32 + hi * 16;
      A.h[0] = *(const v8h*)(rp + k0);
      A.h[1] = *(const v8h*)(rp + k0 + 8);
#pragma unroll
      for (int nb = 0; nb < 4; nb++) {
        v16h Bm = *(const v16h*)(wskipT + (size_t)(nb * 16 + m) * CIN + ks * 32 + hi * 16);
        acc[nb] = wmma_f16(A.v, Bm, acc[nb]);
      }
    }
  }
  float bv[4];
#pragma unroll
  for (int nb = 0; nb < 4; nb++) bv[nb] = b2[nb * 16 + m] + bskip[nb * 16 + m];
#pragma unroll
  for (int nb = 0; nb < 4; nb++)
#pragma unroll
    for (int v = 0; v < 8; v++) {
      int R = 8 * (pbase + v + 8 * hi) + o;
      out[(size_t)R * COUT + nb * 16 + m] = acc[nb][v] + bv[nb];
    }
}

// ---------- workspace layout (bytes, 256-aligned) ----------
static const size_t OFF_GAB1   = 0;                       // 2*128*2 f32
static const size_t OFF_GAB2   = 4096;                    // 2*64*2 f32
static const size_t OFF_TABLE  = 8192;                    // 65536 int
static const size_t OFF_W1AGG  = 270336;                  // 64*64*128 f16
static const size_t OFF_W2T    = 1318912;                 // 27*64*64 f16
static const size_t OFF_WSKIPT = 1540096;                 // 64*128 f16
static const size_t OFF_PART1  = 1556480;                 // 256*128*2 f32
static const size_t OFF_PART2  = 1818624;                 // 512*64*2 f32
static const size_t OFF_F16RAW = 2080768;                 // 32768*128 f16
static const size_t OFF_HCO    = 10469376;                // 32768*128 f16
static const size_t OFF_H1     = 18857984;                // 262144*64 f16
static const size_t OFF_HMID   = 52412416;                // 262144*64 f16

extern "C" void kernel_launch(void* const* d_in, const int* in_sizes, int n_in,
                              void* d_out, int out_size, void* d_ws, size_t ws_size,
                              hipStream_t stream) {
  const float* feats  = (const float*)d_in[0];
  const int*   coords = (const int*)d_in[1];
  const float* gn1_w  = (const float*)d_in[2];
  const float* gn1_b  = (const float*)d_in[3];
  const float* w1     = (const float*)d_in[4];
  const float* b1     = (const float*)d_in[5];
  const float* gn2_w  = (const float*)d_in[6];
  const float* gn2_b  = (const float*)d_in[7];
  const float* w2     = (const float*)d_in[8];
  const float* b2     = (const float*)d_in[9];
  const float* w_skip = (const float*)d_in[10];
  const float* b_skip = (const float*)d_in[11];
  float* out = (float*)d_out;
  char* ws = (char*)d_ws;

  float*     gAB1   = (float*)(ws + OFF_GAB1);
  float*     gAB2   = (float*)(ws + OFF_GAB2);
  int*       table  = (int*)(ws + OFF_TABLE);
  _Float16*  w1aggT = (_Float16*)(ws + OFF_W1AGG);
  _Float16*  w2T    = (_Float16*)(ws + OFF_W2T);
  _Float16*  wskipT = (_Float16*)(ws + OFF_WSKIPT);
  float*     part1  = (float*)(ws + OFF_PART1);
  float*     part2  = (float*)(ws + OFF_PART2);
  _Float16*  f16raw = (_Float16*)(ws + OFF_F16RAW);
  _Float16*  hco    = (_Float16*)(ws + OFF_HCO);
  _Float16*  h1     = (_Float16*)(ws + OFF_H1);
  _Float16*  hmid   = (_Float16*)(ws + OFF_HMID);

  // weight prep
  k_w1agg <<<2048, 256, 0, stream>>>(w1, w1aggT);
  k_w2t   <<<432,  256, 0, stream>>>(w2, w2T);
  k_wskipt<<<32,   256, 0, stream>>>(w_skip, wskipT);
  // coarse voxel table
  k_tinit <<<256,  256, 0, stream>>>(table);
  k_tscat <<<128,  256, 0, stream>>>(coords, table);
  // GN1
  k_gn1_part <<<256, 256, 0, stream>>>(feats, part1);
  k_gn1_final<<<1,   256, 0, stream>>>(part1, gn1_w, gn1_b, gAB1);
  k_apply1   <<<4096, 256, 0, stream>>>(feats, gAB1, f16raw, hco);
  // conv1 (fused subdivide + aggregated 8-tap GEMM)
  k_conv1<<<4096, 128, 0, stream>>>(hco, (const int4*)coords, table, w1aggT, b1, h1);
  // GN2
  k_gn2_part <<<512, 256, 0, stream>>>(h1, part2);
  k_gn2_final<<<1,   128, 0, stream>>>(part2, gn2_w, gn2_b, gAB2);
  k_apply2   <<<16384, 256, 0, stream>>>(h1, gAB2, hmid);
  // conv2 + skip + biases -> output
  k_conv2<<<4096, 128, 0, stream>>>(hmid, f16raw, (const int4*)coords, table,
                                    w2T, wskipT, b2, b_skip, out);
}
